// HierarchizedPLRNN_28166395527566
// MI455X (gfx1250) — matrix operand order repaired and checked by reading, not compiled, not run on hardware
//
#include <hip/hip_runtime.h>

typedef __attribute__((ext_vector_type(16))) __bf16 v16bf;
typedef __attribute__((ext_vector_type(8)))  __bf16 v8bf;
typedef __attribute__((ext_vector_type(8)))  float  v8f;

#define B_    256
#define T_    1024
#define DX    64
#define DZ    128
#define DH    512
#define DS    16
#define ALPHA 0.125f

__device__ __forceinline__ __bf16 bfc(float f) { return (__bf16)f; }

__global__ __launch_bounds__(256, 1)
void plrnn_step_kernel(const float* __restrict__ X,
                       const float* __restrict__ S,
                       const float* __restrict__ A,
                       const float* __restrict__ W1,
                       const float* __restrict__ W2,
                       const float* __restrict__ h1,
                       const float* __restrict__ h2,
                       const float* __restrict__ C,
                       float* __restrict__ out)
{
    const int b    = blockIdx.x;
    const int tid  = threadIdx.x;
    const int wave = tid >> 5;
    const int lane = tid & 31;
    const int lrow = lane & 15;        // A-operand row within 16-row tile
    const int hb   = lane >> 4;        // 0/1: which K-half of a 32-chunk
    const int hk   = hb * 8;

    __shared__ __align__(32) float  z32[DZ];        // recurrent state (fp32)
    __shared__ __align__(32) float  zf32[DZ];       // teacher-forced state (fp32)
    __shared__ __align__(32) __bf16 zfb[DZ];        // zf as bf16 (W2 B-operand)
    __shared__ __align__(32) __bf16 preb[DH + 32];  // [relu(pre) 512 | s_t 16 | zeros 16]
    __shared__ __align__(32) float  h1s[DZ];
    __shared__ __align__(32) float  h2s[DH];
    __shared__ __align__(32) float  As[DZ];

    // ---------------- Load per-batch weights into registers (bf16, WMMA A-layout) ----
    // A-layout (16-bit 16x32): lane L<16 holds row lrow, K = {0..7, 16..23};
    //                          lane L>=16 holds row lrow, K = {8..15, 24..31}.
    v16bf w2a[4][4];   // [tile][kchunk]; tile rows = wave*64 + t*16 + lrow, K = c*32 + ...
    {
        const float* g = W2 + ((size_t)b * DH * DZ);
        #pragma unroll
        for (int t = 0; t < 4; ++t) {
            const float* gr = g + ((size_t)(wave * 64 + t * 16 + lrow)) * DZ;
            #pragma unroll
            for (int c = 0; c < 4; ++c) {
                v16bf a_;
                #pragma unroll
                for (int i = 0; i < 8; ++i) {
                    a_[i]     = bfc(gr[c * 32 + hk + i]);
                    a_[8 + i] = bfc(gr[c * 32 + 16 + hk + i]);
                }
                w2a[t][c] = a_;
            }
        }
    }
    v16bf w1a[17];     // rows = wave*16 + lrow; chunks 0..15 = W1 (K=512), 16 = [C | 0]
    {
        const float* g = W1 + ((size_t)b * DZ * DH) + ((size_t)(wave * 16 + lrow)) * DH;
        #pragma unroll
        for (int c = 0; c < 16; ++c) {
            v16bf a_;
            #pragma unroll
            for (int i = 0; i < 8; ++i) {
                a_[i]     = bfc(g[c * 32 + hk + i]);
                a_[8 + i] = bfc(g[c * 32 + 16 + hk + i]);
            }
            w1a[c] = a_;
        }
        const float* gc = C + ((size_t)b * DZ * DS) + ((size_t)(wave * 16 + lrow)) * DS;
        v16bf a_;
        #pragma unroll
        for (int i = 0; i < 8; ++i) {
            a_[i]     = bfc(gc[hk + i]);   // hk+i in 0..15 -> C columns
            a_[8 + i] = bfc(0.0f);         // K 528..543 padding
        }
        w1a[16] = a_;
    }

    // ---------------- LDS init: biases, A, z0 (teacher force alpha=1 on X[:,0]) ----
    for (int i = tid; i < DH; i += 256) h2s[i] = h2[(size_t)b * DH + i];
    if (tid < DZ) {
        h1s[tid] = h1[(size_t)b * DZ + tid];
        As[tid]  = A[(size_t)b * DZ + tid];
        float z0 = 0.0f;
        if (tid < DX) {
            float x0 = X[(size_t)b * T_ * DX + tid];
            z0 = (x0 != x0) ? 0.0f : x0;
        }
        z32[tid] = z0;
    }
    if (tid < 16) preb[DH + 16 + tid] = bfc(0.0f);   // zero pad of chunk 16

    const float* Xb = X   + (size_t)b * T_ * DX;
    const float* Sb = S   + (size_t)b * T_ * DS;
    float*       Ob = out + (size_t)b * T_ * DX;

    for (int t = 0; t < T_; ++t) {
        __syncthreads();   // previous z32 (or init) visible

        // -------- teacher forcing + stage s_t --------
        if (tid < DZ) {
            float z  = z32[tid];
            float zf = z;
            if (tid < DX) {
                float x = Xb[(size_t)t * DX + tid];
                zf = (x != x) ? z : (ALPHA * x + (1.0f - ALPHA) * z);
            }
            zf32[tid] = zf;
            zfb[tid]  = bfc(zf);
        } else if (tid < DZ + DS) {
            preb[DH + (tid - DZ)] = bfc(Sb[(size_t)t * DS + (tid - DZ)]);
        }
        __syncthreads();

        // -------- pre = W2 @ zf ; relu(pre + h2) -> preb --------
        // B-operand: zf broadcast across all 16 columns -> D replicated per column,
        // so every lane holds the full 16-row tile (rows r + 8*hb in v8f lanes).
        v8f acc[4] = {};
        #pragma unroll
        for (int c = 0; c < 4; ++c) {
            const v8bf bzl = *(const v8bf*)(zfb + c * 32 + hk);
            const v8bf bzh = *(const v8bf*)(zfb + c * 32 + 16 + hk);
            const v16bf bz = __builtin_shufflevector(bzl, bzh, 0, 1, 2, 3, 4, 5, 6, 7, 8, 9, 10, 11, 12, 13, 14, 15);
            #pragma unroll
            for (int ti = 0; ti < 4; ++ti) {
                acc[ti] = __builtin_amdgcn_wmma_f32_16x16x32_bf16(
                    false, w2a[ti][c], false, bz, (short)0, acc[ti], false, false);
            }
        }
        if (lrow == 0) {
            #pragma unroll
            for (int ti = 0; ti < 4; ++ti) {
                const int base = wave * 64 + ti * 16 + hk;   // 8 consecutive rows
                v8bf p;
                #pragma unroll
                for (int r = 0; r < 8; ++r) {
                    float v = acc[ti][r] + h2s[base + r];
                    p[r] = bfc(v > 0.0f ? v : 0.0f);
                }
                *(v8bf*)(preb + base) = p;
            }
        }
        __syncthreads();

        // -------- z' = [W1|C] @ [relu(pre); s_t; 0] + A*zf + h1 --------
        v8f za[4] = {};
        #pragma unroll
        for (int c = 0; c < 17; ++c) {
            const v8bf bpl = *(const v8bf*)(preb + c * 32 + hk);
            const v8bf bph = *(const v8bf*)(preb + c * 32 + 16 + hk);
            const v16bf bp = __builtin_shufflevector(bpl, bph, 0, 1, 2, 3, 4, 5, 6, 7, 8, 9, 10, 11, 12, 13, 14, 15);
            za[c & 3] = __builtin_amdgcn_wmma_f32_16x16x32_bf16(
                false, w1a[c], false, bp, (short)0, za[c & 3], false, false);
        }
        v8f zt = (za[0] + za[1]) + (za[2] + za[3]);
        if (lrow == 0) {
            const int base = wave * 16 + hk;                 // 8 consecutive z rows
            float zn[8];
            #pragma unroll
            for (int r = 0; r < 8; ++r) {
                const int m = base + r;
                zn[r]  = zt[r] + As[m] * zf32[m] + h1s[m];
                z32[m] = zn[r];
            }
            if (base < DX) {                                 // emit Z[:, t, :64]
                float* o = Ob + (size_t)t * DX + base;
                #pragma unroll
                for (int r = 0; r < 8; ++r) o[r] = zn[r];
            }
        }
    }
}

extern "C" void kernel_launch(void* const* d_in, const int* in_sizes, int n_in,
                              void* d_out, int out_size, void* d_ws, size_t ws_size,
                              hipStream_t stream) {
    (void)in_sizes; (void)n_in; (void)out_size; (void)d_ws; (void)ws_size;
    const float* X  = (const float*)d_in[0];
    const float* S  = (const float*)d_in[1];
    const float* A  = (const float*)d_in[2];
    const float* W1 = (const float*)d_in[3];
    const float* W2 = (const float*)d_in[4];
    const float* h1 = (const float*)d_in[5];
    const float* h2 = (const float*)d_in[6];
    const float* C  = (const float*)d_in[7];
    float* out = (float*)d_out;

    plrnn_step_kernel<<<B_, 256, 0, stream>>>(X, S, A, W1, W2, h1, h2, C, out);
}
